// VanillaTransformerModel_29188597743869
// MI455X (gfx1250) — matrix operand order repaired and checked by reading, not compiled, not run on hardware
//
#include <hip/hip_runtime.h>
#include <math.h>
#include <stdint.h>

// ---------------- model constants ----------------
#define LNUM 4
#define CDIM 768
#define NHEAD 12
#define TDIM 1024
#define BDIM 2
#define VDIM 32000
#define HSZ 64
#define BT (BDIM * TDIM)

// ---------------- WMMA / descriptor types ----------------
typedef __attribute__((ext_vector_type(16))) __bf16 v16bf;
typedef __attribute__((ext_vector_type(8))) float v8f;
typedef __attribute__((ext_vector_type(4))) unsigned int u32x4;
typedef __attribute__((ext_vector_type(8))) unsigned int u32x8;

union FragAB {
  uint4 q[2];
  v16bf v;
};

// epilogue modes
enum { EPI_F32 = 0, EPI_BF16 = 1, EPI_GELU_BF16 = 2, EPI_ADD_F32 = 3 };

// =====================================================================
// Generic bf16 WMMA GEMM:  D[m,n] = scale * sum_k A[m,k]*Bt[n,k] + bias[n]
// A: row-major [M,K] (row stride lda), Bt: n-major [N,K] (row stride ldb).
// Block = 256 threads = 8 waves; tile 128x128, K-step 32.
// A-tile: global_load_async_to_lds_b128 (ASYNCcnt, per-lane DMA).
// B-tile: TDM tensor_load_to_lds (TENSORcnt), one 32x16 sub-tile per wave,
//         with hardware LDS padding (16 DWORDs data + 4 DWORDs pad = 80B row)
//         matching the LDR=40 fragment layout.
// Double-buffered ping-pong with one split-barrier per k-step.
// Wave (wm 0..3, wn 0..1) owns 32x64 -> 2x4 accumulators of 16x16.
// Requires: M%128==0, N%128==0, K%32==0, lda/ldb multiples of 8.
// =====================================================================
template <int EPI>
__global__ __launch_bounds__(256) void gemm_bf16_wmma(
    const __bf16* __restrict__ A, long long strideAz, int lda,
    const __bf16* __restrict__ Bt, long long strideBz, int ldb,
    const float* __restrict__ bias, void* __restrict__ Dvoid,
    long long strideDz, int ldd, int M, int N, int K, float scale) {
  constexpr int LDR = 40;  // padded LDS row stride (bf16): 80B -> conflict-free
  __shared__ __align__(16) __bf16 shA[2][128 * LDR];
  __shared__ __align__(16) __bf16 shB[2][128 * LDR];
  constexpr unsigned BUFB = 128 * LDR * 2;  // bytes per LDS buffer
  constexpr unsigned ROW64 = 64 * LDR * 2;  // byte offset of chunk-1 rows

  const int tid = threadIdx.x;
  const int lane = tid & 31;
  const int wave = tid >> 5;
  const int wm = wave & 3;   // 4 row-chunks of 32
  const int wn = wave >> 2;  // 2 col-chunks of 64
  const int hl = lane >> 4;  // lane half
  const int r = lane & 15;

  const int z = blockIdx.z;
  A += (long long)z * strideAz;
  Bt += (long long)z * strideBz;

  const int m0 = blockIdx.y * 128;
  const int n0 = blockIdx.x * 128;

  // ---- A-tile async copy coordinates: 512 chunks of 8 bf16, 2 per thread ----
  const int rowc = tid >> 2;
  const int kq = (tid & 3) << 3;
  const long long gA0 = (long long)(m0 + rowc) * lda + kq;
  const long long gA1 = (long long)(m0 + rowc + 64) * lda + kq;
  const unsigned ldsA = (unsigned)(uintptr_t)&shA[0][rowc * LDR + kq];

  auto issue_tileA = [&](int k0, int bsel) {
    const unsigned la = ldsA + (unsigned)bsel * BUFB;
    asm volatile("global_load_async_to_lds_b128 %0, %1, off" ::"v"(la),
                 "v"((unsigned long long)(uintptr_t)(A + gA0 + k0))
                 : "memory");
    asm volatile("global_load_async_to_lds_b128 %0, %1, off" ::"v"(la + ROW64),
                 "v"((unsigned long long)(uintptr_t)(A + gA1 + k0))
                 : "memory");
  };

  // ---- B-tile TDM: each wave DMAs rows n = wq*16 .. wq*16+15 of the tile ----
  const int wq = __builtin_amdgcn_readfirstlane(wave);  // wave-uniform SGPR
  const unsigned ldsBbase = (unsigned)(uintptr_t)&shB[0][0];
  auto issue_tileB = [&](int k0, int bsel) {
    const unsigned lds =
        ldsBbase + (unsigned)bsel * BUFB + (unsigned)wq * 16u * (unsigned)(LDR * 2);
    const unsigned long long ga = (unsigned long long)(uintptr_t)(
        Bt + (long long)(n0 + wq * 16) * ldb + k0);
    u32x4 g0;
    g0.x = 1u;                                  // count=1 valid user descriptor
    g0.y = lds;                                 // lds_addr (bytes)
    g0.z = (unsigned)ga;                        // global_addr[31:0]
    g0.w = (unsigned)(ga >> 32) | (2u << 30);   // global_addr[56:32] | type=2
    u32x8 g1;
    g1[0] = (1u << 16)      // data_size = 2 bytes (bf16)
            | (1u << 20)    // pad_enable
            | (3u << 22)    // pad_interval: every 16 DWORDs (row = 32 bf16)
            | (3u << 25);   // pad_amount: 4 DWORDs -> 80B row stride (LDR=40)
    g1[1] = 32u << 16;      // tensor_dim0 = 32 (tile-sized tensor: no OOB)
    g1[2] = 16u << 16;      // tensor_dim0 hi=0 | tensor_dim1 = 16
    g1[3] = 32u << 16;      // tensor_dim1 hi=0 | tile_dim0 = 32
    g1[4] = 16u;            // tile_dim1 = 16, tile_dim2 = 0
    g1[5] = (unsigned)ldb;  // tensor_dim0_stride[31:0] (elements)
    g1[6] = 0u;             // stride hi | dim1_stride lo
    g1[7] = 0u;
    asm volatile("tensor_load_to_lds %0, %1" ::"s"(g0), "s"(g1) : "memory");
  };

  v8f acc[2][4] = {};

  // prologue: stage tile 0
  issue_tileA(0, 0);
  issue_tileB(0, 0);
  asm volatile("s_wait_asynccnt 0x0" ::: "memory");
  __builtin_amdgcn_s_wait_tensorcnt(0);
  __syncthreads();

  int buf = 0;
  for (int k0 = 0; k0 < K; k0 += 32) {
    const bool more = (k0 + 32 < K);
    // overlap: DMA next tile into the other buffer while computing this one
    if (more) {
      issue_tileA(k0 + 32, buf ^ 1);
      issue_tileB(k0 + 32, buf ^ 1);
    }
    if (k0 + 64 < K) {  // L2 prefetch two tiles ahead (global_prefetch_b8)
      __builtin_prefetch(A + gA0 + k0 + 64, 0, 1);
      __builtin_prefetch(Bt + gA0 + k0 + 64, 0, 1);
    }

    // ---- fragments (ISA 16-bit 16x32 layout) ----
    // A lane: row = r; K chunks at 8*hl..8*hl+7 and 16+8*hl..23+8*hl
    FragAB fa[2], fb[4];
#pragma unroll
    for (int i = 0; i < 2; i++) {
      const int row = wm * 32 + i * 16 + r;
      fa[i].q[0] = *(const uint4*)(&shA[buf][row * LDR + 8 * hl]);
      fa[i].q[1] = *(const uint4*)(&shA[buf][row * LDR + 16 + 8 * hl]);
    }
    // B lane: col = r; K chunks at 8*hl..8*hl+7 and 16+8*hl..23+8*hl
#pragma unroll
    for (int j = 0; j < 4; j++) {
      const int col = wn * 64 + j * 16 + r;
      fb[j].q[0] = *(const uint4*)(&shB[buf][col * LDR + 8 * hl]);
      fb[j].q[1] = *(const uint4*)(&shB[buf][col * LDR + 16 + 8 * hl]);
    }
#pragma unroll
    for (int i = 0; i < 2; i++)
#pragma unroll
      for (int j = 0; j < 4; j++)
        acc[i][j] = __builtin_amdgcn_wmma_f32_16x16x32_bf16(
            false, fa[i].v, false, fb[j].v, (short)0, acc[i][j], false, false);

    if (more) {
      // next tile fully in LDS, then block-wide rendezvous before flipping
      asm volatile("s_wait_asynccnt 0x0" ::: "memory");
      __builtin_amdgcn_s_wait_tensorcnt(0);
      __syncthreads();
      buf ^= 1;
    }
  }

  // ---- epilogue: D layout lane(r)=col, vgpr e -> row 8*hl+e ----
#pragma unroll
  for (int i = 0; i < 2; i++) {
    const int mbase = m0 + wm * 32 + i * 16 + hl * 8;
#pragma unroll
    for (int j = 0; j < 4; j++) {
      const int n = n0 + wn * 64 + j * 16 + r;
      const float bv = bias ? bias[n] : 0.0f;
#pragma unroll
      for (int e = 0; e < 8; e++) {
        float v = acc[i][j][e] * scale + bv;
        const long long didx = (long long)(mbase + e) * ldd + n;
        if constexpr (EPI == EPI_F32) {
          (((float*)Dvoid) + (long long)z * strideDz)[didx] = v;
        } else if constexpr (EPI == EPI_ADD_F32) {
          float* Df = ((float*)Dvoid) + (long long)z * strideDz;
          Df[didx] += v;
        } else if constexpr (EPI == EPI_GELU_BF16) {
          v = 0.5f * v * (1.0f + erff(v * 0.70710678118654752f));  // exact erf
          (((__bf16*)Dvoid) + (long long)z * strideDz)[didx] = (__bf16)v;
        } else {  // EPI_BF16
          (((__bf16*)Dvoid) + (long long)z * strideDz)[didx] = (__bf16)v;
        }
      }
    }
  }
}

// =====================================================================
// Block-wide reductions (256 threads = 8 waves, wave32)
// =====================================================================
__device__ __forceinline__ float block_reduce_add(float v) {
  for (int m = 16; m; m >>= 1) v += __shfl_xor(v, m, 32);
  __shared__ float sh[8];
  const int lane = threadIdx.x & 31, w = threadIdx.x >> 5;
  __syncthreads();
  if (lane == 0) sh[w] = v;
  __syncthreads();
  float r = 0.f;
#pragma unroll
  for (int i = 0; i < 8; i++) r += sh[i];
  return r;
}

__device__ __forceinline__ float block_reduce_max(float v) {
  for (int m = 16; m; m >>= 1) v = fmaxf(v, __shfl_xor(v, m, 32));
  __shared__ float sh[8];
  const int lane = threadIdx.x & 31, w = threadIdx.x >> 5;
  __syncthreads();
  if (lane == 0) sh[w] = v;
  __syncthreads();
  float r = -INFINITY;
#pragma unroll
  for (int i = 0; i < 8; i++) r = fmaxf(r, sh[i]);
  return r;
}

// =====================================================================
// Small kernels
// =====================================================================
__global__ __launch_bounds__(256) void embed_kernel(
    const int* __restrict__ ids, const float* __restrict__ wte,
    const float* __restrict__ wpe, float* __restrict__ x) {
  const int token = blockIdx.x;
  const int t = token % TDIM;
  const int id = ids[token];
#pragma unroll
  for (int i = 0; i < 3; i++) {
    const int c = threadIdx.x + i * 256;
    x[(long long)token * CDIM + c] =
        wte[(long long)id * CDIM + c] + wpe[(long long)t * CDIM + c];
  }
}

// LayerNorm over C=768; writes bf16 row-major; optionally also transposed
// bf16 [B][C][T] (n-major operand for the att @ ln GEMM).
__global__ __launch_bounds__(256) void layernorm_kernel(
    const float* __restrict__ x, const float* __restrict__ w,
    const float* __restrict__ b, __bf16* __restrict__ out,
    __bf16* __restrict__ outT) {
  const int token = blockIdx.x;
  const int bi = token / TDIM, t = token % TDIM;
  float vals[3];
  float s = 0.f;
#pragma unroll
  for (int i = 0; i < 3; i++) {
    vals[i] = x[(long long)token * CDIM + threadIdx.x + i * 256];
    s += vals[i];
  }
  const float mean = block_reduce_add(s) * (1.0f / CDIM);
  float s2 = 0.f;
#pragma unroll
  for (int i = 0; i < 3; i++) {
    const float d = vals[i] - mean;
    s2 += d * d;
  }
  const float var = block_reduce_add(s2) * (1.0f / CDIM);
  const float rstd = rsqrtf(var + 1e-5f);
#pragma unroll
  for (int i = 0; i < 3; i++) {
    const int c = threadIdx.x + i * 256;
    const float nv = (vals[i] - mean) * rstd * w[c] + b[c];
    out[(long long)token * CDIM + c] = (__bf16)nv;
    if (outT) outT[((long long)bi * CDIM + c) * TDIM + t] = (__bf16)nv;
  }
}

// Causal softmax over one row of [T] scores; writes bf16 probabilities.
__global__ __launch_bounds__(256) void softmax_causal_kernel(
    const float* __restrict__ scores, __bf16* __restrict__ att) {
  const long long row = blockIdx.x;  // (b*NH+h)*T + t
  const int t = (int)(row % TDIM);
  const float* srow = scores + row * TDIM;
  __bf16* arow = att + row * TDIM;
  float mx = -INFINITY;
  for (int j = threadIdx.x; j < TDIM; j += 256)
    if (j <= t) mx = fmaxf(mx, srow[j]);
  mx = block_reduce_max(mx);
  float s = 0.f;
  for (int j = threadIdx.x; j < TDIM; j += 256)
    if (j <= t) s += expf(srow[j] - mx);
  s = block_reduce_add(s);
  const float inv = 1.0f / s;
  for (int j = threadIdx.x; j < TDIM; j += 256) {
    const float e = (j <= t) ? expf(srow[j] - mx) * inv : 0.0f;
    arow[j] = (__bf16)e;
  }
}

// x[b,t,c] += sum_h attended[b,h,t,c] * hw[h,c]
__global__ __launch_bounds__(256) void head_reduce_kernel(
    const float* __restrict__ attended, const float* __restrict__ hw,
    float* __restrict__ x) {
  const int token = blockIdx.x;
  const int bi = token / TDIM, t = token % TDIM;
#pragma unroll
  for (int i = 0; i < 3; i++) {
    const int c = threadIdx.x + i * 256;
    float acc = 0.f;
#pragma unroll
    for (int h = 0; h < NHEAD; h++)
      acc += attended[(((long long)(bi * NHEAD + h)) * TDIM + t) * CDIM + c] *
             hw[h * CDIM + c];
    x[(long long)token * CDIM + c] += acc;
  }
}

__global__ __launch_bounds__(256) void convert_bf16_kernel(
    const float* __restrict__ in, __bf16* __restrict__ out, long long n) {
  const long long i = (long long)blockIdx.x * 256 + threadIdx.x;
  if (i < n) out[i] = (__bf16)in[i];
}

// in: [K][N] f32 row-major -> out: [N][K] bf16 (n-major B operand)
__global__ __launch_bounds__(256) void transpose_bf16_kernel(
    const float* __restrict__ in, __bf16* __restrict__ out, int K, int N) {
  const long long i = (long long)blockIdx.x * 256 + threadIdx.x;
  if (i < (long long)K * N) {
    const int k = (int)(i / N), n = (int)(i % N);
    out[(long long)n * K + k] = (__bf16)in[i];
  }
}

// =====================================================================
// Host orchestration
// =====================================================================
extern "C" void kernel_launch(void* const* d_in, const int* in_sizes, int n_in,
                              void* d_out, int out_size, void* d_ws,
                              size_t ws_size, hipStream_t stream) {
  const int* ids = (const int*)d_in[0];
  const float* wte = (const float*)d_in[1];
  const float* wpe = (const float*)d_in[2];
  const float* ln1_w = (const float*)d_in[3];
  const float* ln1_b = (const float*)d_in[4];
  const float* attn_W = (const float*)d_in[5];
  const float* attn_b = (const float*)d_in[6];
  const float* head_w = (const float*)d_in[7];
  const float* ln2_w = (const float*)d_in[8];
  const float* ln2_b = (const float*)d_in[9];
  const float* fc_W = (const float*)d_in[10];
  const float* fc_b = (const float*)d_in[11];
  const float* proj_W = (const float*)d_in[12];
  const float* proj_b = (const float*)d_in[13];
  const float* lnf_w = (const float*)d_in[14];
  const float* lnf_b = (const float*)d_in[15];
  float* logits = (float*)d_out;
  (void)in_sizes; (void)n_in; (void)out_size; (void)ws_size;

  // workspace bump allocator
  char* base = (char*)d_ws;
  size_t off = 0;
  auto alloc = [&](size_t bytes) -> void* {
    void* p = base + off;
    off += (bytes + 255) & ~(size_t)255;
    return p;
  };

  float* x = (float*)alloc((size_t)BT * CDIM * 4);
  __bf16* ln = (__bf16*)alloc((size_t)BT * CDIM * 2);
  __bf16* lnT = (__bf16*)alloc((size_t)BDIM * CDIM * TDIM * 2);
  __bf16* qk = (__bf16*)alloc((size_t)BT * 2 * CDIM * 2);
  float* scores = (float*)alloc((size_t)BDIM * NHEAD * TDIM * TDIM * 4);
  __bf16* att = (__bf16*)alloc((size_t)BDIM * NHEAD * TDIM * TDIM * 2);
  float* attended = (float*)alloc((size_t)BDIM * NHEAD * TDIM * CDIM * 4);
  __bf16* h1 = (__bf16*)alloc((size_t)BT * 4 * CDIM * 2);
  __bf16* wteb = (__bf16*)alloc((size_t)VDIM * CDIM * 2);
  __bf16* attnWt = (__bf16*)alloc((size_t)LNUM * 2 * CDIM * CDIM * 2);
  __bf16* fcWt = (__bf16*)alloc((size_t)LNUM * 4 * CDIM * CDIM * 2);
  __bf16* projWt = (__bf16*)alloc((size_t)LNUM * CDIM * 4 * CDIM * 2);

  const dim3 blk(256);

  // one-time weight precision/layout conversion
  {
    const long long n = (long long)VDIM * CDIM;
    convert_bf16_kernel<<<dim3((unsigned)((n + 255) / 256)), blk, 0, stream>>>(
        wte, wteb, n);
  }
  for (int l = 0; l < LNUM; l++) {
    {
      const long long n = (long long)CDIM * 2 * CDIM;
      transpose_bf16_kernel<<<dim3((unsigned)((n + 255) / 256)), blk, 0, stream>>>(
          attn_W + (long long)l * n, attnWt + (long long)l * n, CDIM, 2 * CDIM);
    }
    {
      const long long n = (long long)CDIM * 4 * CDIM;
      transpose_bf16_kernel<<<dim3((unsigned)((n + 255) / 256)), blk, 0, stream>>>(
          fc_W + (long long)l * n, fcWt + (long long)l * n, CDIM, 4 * CDIM);
    }
    {
      const long long n = (long long)4 * CDIM * CDIM;
      transpose_bf16_kernel<<<dim3((unsigned)((n + 255) / 256)), blk, 0, stream>>>(
          proj_W + (long long)l * n, projWt + (long long)l * n, 4 * CDIM, CDIM);
    }
  }

  embed_kernel<<<dim3(BT), blk, 0, stream>>>(ids, wte, wpe, x);

  for (int l = 0; l < LNUM; l++) {
    // ln1 -> bf16 (row-major) + transposed (used as V operand)
    layernorm_kernel<<<dim3(BT), blk, 0, stream>>>(x, ln1_w + l * CDIM,
                                                   ln1_b + l * CDIM, ln, lnT);
    // qk = ln1 @ attn_W + attn_b   [2048 x 1536], bf16 out
    gemm_bf16_wmma<EPI_BF16><<<dim3(2 * CDIM / 128, BT / 128, 1), blk, 0, stream>>>(
        ln, 0, CDIM, attnWt + (long long)l * 2 * CDIM * CDIM, 0, CDIM,
        attn_b + l * 2 * CDIM, qk, 0, 2 * CDIM, BT, 2 * CDIM, CDIM, 1.0f);
    // scores = q @ k^T * 1/sqrt(HS), per (b,h)
    for (int b = 0; b < BDIM; b++) {
      gemm_bf16_wmma<EPI_F32><<<dim3(TDIM / 128, TDIM / 128, NHEAD), blk, 0, stream>>>(
          qk + (long long)b * TDIM * 2 * CDIM, HSZ, 2 * CDIM,
          qk + (long long)b * TDIM * 2 * CDIM + CDIM, HSZ, 2 * CDIM, nullptr,
          scores + (long long)b * NHEAD * TDIM * TDIM, (long long)TDIM * TDIM,
          TDIM, TDIM, TDIM, HSZ, 0.125f);
    }
    softmax_causal_kernel<<<dim3(BDIM * NHEAD * TDIM), blk, 0, stream>>>(scores, att);
    // attended = att @ ln1   (V = full-width ln1), per (b,h)
    for (int b = 0; b < BDIM; b++) {
      gemm_bf16_wmma<EPI_F32><<<dim3(CDIM / 128, TDIM / 128, NHEAD), blk, 0, stream>>>(
          att + (long long)b * NHEAD * TDIM * TDIM, (long long)TDIM * TDIM, TDIM,
          lnT + (long long)b * CDIM * TDIM, 0, TDIM, nullptr,
          attended + (long long)b * NHEAD * TDIM * CDIM, (long long)TDIM * CDIM,
          CDIM, TDIM, CDIM, TDIM, 1.0f);
    }
    // x += sum_h attended * head_w
    head_reduce_kernel<<<dim3(BT), blk, 0, stream>>>(
        attended, head_w + (long long)l * NHEAD * CDIM, x);
    // MLP
    layernorm_kernel<<<dim3(BT), blk, 0, stream>>>(x, ln2_w + l * CDIM,
                                                   ln2_b + l * CDIM, ln, nullptr);
    gemm_bf16_wmma<EPI_GELU_BF16><<<dim3(4 * CDIM / 128, BT / 128, 1), blk, 0, stream>>>(
        ln, 0, CDIM, fcWt + (long long)l * 4 * CDIM * CDIM, 0, CDIM,
        fc_b + l * 4 * CDIM, h1, 0, 4 * CDIM, BT, 4 * CDIM, CDIM, 1.0f);
    gemm_bf16_wmma<EPI_ADD_F32><<<dim3(CDIM / 128, BT / 128, 1), blk, 0, stream>>>(
        h1, 0, 4 * CDIM, projWt + (long long)l * CDIM * 4 * CDIM, 0, 4 * CDIM,
        proj_b + l * CDIM, x, 0, CDIM, BT, CDIM, 4 * CDIM, 1.0f);
  }

  // final LN + tied lm_head (Bt = wte itself, already n-major)
  layernorm_kernel<<<dim3(BT), blk, 0, stream>>>(x, lnf_w, lnf_b, ln, nullptr);
  gemm_bf16_wmma<EPI_F32><<<dim3(VDIM / 128, BT / 128, 1), blk, 0, stream>>>(
      ln, 0, CDIM, wteb, 0, CDIM, nullptr, logits, 0, VDIM, BT, VDIM, CDIM, 1.0f);
}
